// TransformerEncoderBlock_8297876815914
// MI455X (gfx1250) — matrix operand order repaired and checked by reading, not compiled, not run on hardware
//
#include <hip/hip_runtime.h>
#include <hip/hip_bf16.h>
#include <math.h>

// ---------------- types ----------------
typedef __bf16 bf16;
typedef __attribute__((ext_vector_type(16))) __bf16 v16bf;
typedef __attribute__((ext_vector_type(8)))  __bf16 v8bf;
typedef __attribute__((ext_vector_type(8)))  float   v8f;

// ---------------- problem constants ----------------
#define BATCH   32
#define NTOK    577
#define TOKENS  (BATCH*NTOK)          // 18464
#define DMODEL  768
#define NHEAD   12
#define HDIM    32                    // half-head dim
#define MLPD    3072
#define KEYPAD  608                   // 19*32, padded key count
#define VT_ROW  (BATCH*KEYPAD)        // 19456 elements per Vt channel row
#define QTILES  37                    // ceil(577/16)
#define LAMB0   0.2f
#define QSCALE  0.1767766952966369f   // 32^-0.5

// ---------------- helpers ----------------
static __device__ __forceinline__ v8f wmma_bf16(v16bf a, v16bf b, v8f c) {
  // D = A(16x32) * B(32x16) + C, f32 accumulate
  return __builtin_amdgcn_wmma_f32_16x16x32_bf16(false, a, false, b, (short)0, c,
                                                 false, false);
}

// two aligned 16-byte loads -> one 16-element bf16 fragment
static __device__ __forceinline__ v16bf ld16(const bf16* lo, const bf16* hi) {
  v8bf l = *(const v8bf*)lo;
  v8bf h = *(const v8bf*)hi;
  v16bf r;
#pragma unroll
  for (int i = 0; i < 8; ++i) { r[i] = l[i]; r[i + 8] = h[i]; }
  return r;
}

static __device__ __forceinline__ float wred_sum(float v) {
#pragma unroll
  for (int m = 16; m >= 1; m >>= 1) v += __shfl_xor(v, m, 32);
  return v;
}

// ---------------- f32 -> bf16 weight conversion ----------------
__global__ __launch_bounds__(256) void cvt_bf16_kernel(const float* __restrict__ src,
                                                       bf16* __restrict__ dst, int n) {
  int i = blockIdx.x * 256 + threadIdx.x;
  if (i < n) dst[i] = (bf16)src[i];
}

// ---------------- LayerNorm (f32 in, bf16 out), one wave per token ----------------
__global__ __launch_bounds__(256) void layernorm_kernel(const float* __restrict__ x,
                                                        const float* __restrict__ g,
                                                        const float* __restrict__ b,
                                                        bf16* __restrict__ out) {
  const int t    = blockIdx.x * 8 + (threadIdx.x >> 5);
  const int lane = threadIdx.x & 31;
  if (t >= TOKENS) return;
  const float* xr = x + (size_t)t * DMODEL;
  float v[24];
  float s = 0.f;
#pragma unroll
  for (int i = 0; i < 24; ++i) { v[i] = xr[lane + 32 * i]; s += v[i]; }
  s = wred_sum(s);
  const float mu = s * (1.f / DMODEL);
  float q = 0.f;
#pragma unroll
  for (int i = 0; i < 24; ++i) { float d = v[i] - mu; q += d * d; }
  q = wred_sum(q);
  const float inv = rsqrtf(q * (1.f / DMODEL) + 1e-5f);
  bf16* o = out + (size_t)t * DMODEL;
#pragma unroll
  for (int i = 0; i < 24; ++i) {
    int c = lane + 32 * i;
    o[c] = (bf16)((v[i] - mu) * inv * g[c] + b[c]);
  }
}

// ---------------- WMMA GEMM: C[T,N] = A[T,K] @ W[N,K]^T, fused epilogues ----------------
// mode 0: outB[m*N+n]  = bf16(alpha * c)                       (Q / K projection)
// mode 1: outB[n*VT_ROW + b*KEYPAD + m%577] = bf16(c)          (V, transposed+padded)
// mode 2: outF[m*N+n]  = c + resid[m*N+n]                      (attn out proj + residual)
// mode 3: outB[m*N+n]  = bf16(gelu(c + bias[n]))               (MLP up)
// mode 4: outF[m*N+n]  = c + bias[n] + resid[m*N+n]            (MLP down + residual)
__global__ __launch_bounds__(256) void wmma_gemm_kernel(
    const bf16* __restrict__ A, const bf16* __restrict__ W, int K, int N, int mode,
    float alpha, const float* __restrict__ resid, const float* __restrict__ bias,
    float* __restrict__ outF, bf16* __restrict__ outB) {
  const int lane = threadIdx.x & 31;
  const int wave = threadIdx.x >> 5;
  const int g    = lane >> 4;     // half-wave group
  const int l15  = lane & 15;
  const int m0   = blockIdx.y * 256 + wave * 32;   // 32 rows per wave
  const int n0   = blockIdx.x * 64;                // 64 cols per wave

  v8f acc[2][4];
#pragma unroll
  for (int r = 0; r < 2; ++r)
#pragma unroll
    for (int nt = 0; nt < 4; ++nt) acc[r][nt] = (v8f){};

  size_t rowA[2];
#pragma unroll
  for (int r = 0; r < 2; ++r) {
    int m = m0 + r * 16 + l15;
    if (m >= TOKENS) m = TOKENS - 1;               // clamp: pad rows read row T-1
    rowA[r] = (size_t)m * K;
  }

  for (int k0 = 0; k0 < K; k0 += 32) {
    v16bf a[2];
#pragma unroll
    for (int r = 0; r < 2; ++r)
      a[r] = ld16(A + rowA[r] + k0 + g * 16, A + rowA[r] + k0 + g * 16 + 8);
#pragma unroll
    for (int nt = 0; nt < 4; ++nt) {
      const bf16* wp = W + (size_t)(n0 + nt * 16 + l15) * K + k0 + g * 16;
      v16bf bfr = ld16(wp, wp + 8);
      acc[0][nt] = wmma_bf16(a[0], bfr, acc[0][nt]);
      acc[1][nt] = wmma_bf16(a[1], bfr, acc[1][nt]);
    }
    if (k0 + 32 < K) {   // gfx1250 global_prefetch_b8 of next K-slab
      __builtin_prefetch(A + rowA[0] + k0 + 32, 0, 0);
      __builtin_prefetch(W + (size_t)(n0 + l15) * K + k0 + 32, 0, 0);
    }
  }

#pragma unroll
  for (int r = 0; r < 2; ++r)
#pragma unroll
    for (int nt = 0; nt < 4; ++nt)
#pragma unroll
      for (int j = 0; j < 8; ++j) {
        const int m = m0 + r * 16 + j + 8 * g;
        if (m >= TOKENS) continue;
        const int n = n0 + nt * 16 + l15;
        float v = acc[r][nt][j] * alpha;
        if (mode == 0) {
          outB[(size_t)m * N + n] = (bf16)v;
        } else if (mode == 1) {
          int bb = m / NTOK;
          int ml = m - bb * NTOK;
          outB[(size_t)n * VT_ROW + bb * KEYPAD + ml] = (bf16)v;
        } else if (mode == 2) {
          outF[(size_t)m * N + n] = v + resid[(size_t)m * N + n];
        } else if (mode == 3) {
          float t  = v + bias[n];
          float ge = 0.5f * t * (1.f + erff(t * 0.70710678118654752f));
          outB[(size_t)m * N + n] = (bf16)ge;
        } else {
          outF[(size_t)m * N + n] = v + bias[n] + resid[(size_t)m * N + n];
        }
      }
}

// ---------------- Differential flash attention, fully in WMMA, zero LDS ----------------
// One wave per (batch, head, 16-query tile). Computes S^T = K * Q^T so that the
// probability C-fragments repack in-lane into the next WMMA's A-fragment.
__global__ __launch_bounds__(256) void diff_attn_kernel(
    const bf16* __restrict__ Q, const bf16* __restrict__ Kb, const bf16* __restrict__ Vt,
    const float* __restrict__ lq1, const float* __restrict__ lk1,
    const float* __restrict__ lq2, const float* __restrict__ lk2,
    const float* __restrict__ subg, bf16* __restrict__ Ao) {
  const int lane = threadIdx.x & 31;
  const int g    = lane >> 4;
  const int l15  = lane & 15;
  const int job  = blockIdx.x * 8 + (threadIdx.x >> 5);
  const int b    = job / (NHEAD * QTILES);
  const int h    = (job / QTILES) % NHEAD;
  const int q0   = (job % QTILES) * 16;

  // lambda = exp(sum lq1*lk1) - exp(sum lq2*lk2) + 0.2   (HDIM==32 -> one elem/lane)
  const float s1sum = wred_sum(lq1[lane] * lk1[lane]);
  const float s2sum = wred_sum(lq2[lane] * lk2[lane]);
  const float lam   = __expf(s1sum) - __expf(s2sum) + LAMB0;

  // Q^T B-fragments, resident for the whole job (scale already folded into Qb)
  int qrow = q0 + l15;
  if (qrow > NTOK - 1) qrow = NTOK - 1;
  const bf16* qp = Q + (size_t)(b * NTOK + qrow) * DMODEL + h * 64 + g * 16;
  const v16bf bq1 = ld16(qp, qp + 8);
  const v16bf bq2 = ld16(qp + 32, qp + 40);

  v8f o1[4], o2[4];
#pragma unroll
  for (int nt = 0; nt < 4; ++nt) { o1[nt] = (v8f){}; o2[nt] = (v8f){}; }
  float m1 = -1e30f, m2 = -1e30f, l1 = 0.f, l2 = 0.f;

  for (int mc = 0; mc < 19; ++mc) {          // 19 chunks of 32 keys (608 padded)
    v8f s1[2], s2[2];
#pragma unroll
    for (int t = 0; t < 2; ++t) {            // two 16-key tiles: S^T = K_tile * Q^T
      int key = mc * 32 + t * 16 + l15;
      if (key > NTOK - 1) key = NTOK - 1;
      const bf16* kp = Kb + (size_t)(b * NTOK + key) * DMODEL + h * 64;
      v16bf ak1 = ld16(kp + g * 16, kp + 8 + g * 16);
      v16bf ak2 = ld16(kp + 32 + g * 16, kp + 40 + g * 16);
      v8f z = (v8f){};
      s1[t] = wmma_bf16(ak1, bq1, z);
      s2[t] = wmma_bf16(ak2, bq2, z);
    }

    // mask padded keys, per-column (query) chunk max
    float cm1 = -1e30f, cm2 = -1e30f;
#pragma unroll
    for (int j = 0; j < 8; ++j) {
      const int k0v = mc * 32 + 8 * g + j;       // tile0 key for C-VGPR j
      const int k1v = k0v + 16;                  // tile1 key
      if (k0v >= NTOK) { s1[0][j] = -1e30f; s2[0][j] = -1e30f; }
      if (k1v >= NTOK) { s1[1][j] = -1e30f; s2[1][j] = -1e30f; }
      cm1 = fmaxf(cm1, fmaxf(s1[0][j], s1[1][j]));
      cm2 = fmaxf(cm2, fmaxf(s2[0][j], s2[1][j]));
    }
    cm1 = fmaxf(cm1, __shfl_xor(cm1, 16, 32));
    cm2 = fmaxf(cm2, __shfl_xor(cm2, 16, 32));
    const float mn1 = fmaxf(m1, cm1), mn2 = fmaxf(m2, cm2);
    const float c1 = __expf(m1 - mn1), c2 = __expf(m2 - mn2);
    m1 = mn1; m2 = mn2;

    // probabilities: C-frag -> A-frag is a pure in-lane repack on CDNA5
    float rs1 = 0.f, rs2 = 0.f;
    v16bf ap1, ap2;
#pragma unroll
    for (int j = 0; j < 8; ++j) {
      float p0 = __expf(s1[0][j] - m1);
      float p1 = __expf(s1[1][j] - m1);
      float r0 = __expf(s2[0][j] - m2);
      float r1 = __expf(s2[1][j] - m2);
      rs1 += p0 + p1; rs2 += r0 + r1;
      ap1[j] = (bf16)p0; ap1[j + 8] = (bf16)p1;
      ap2[j] = (bf16)r0; ap2[j + 8] = (bf16)r1;
    }
    rs1 += __shfl_xor(rs1, 16, 32);
    rs2 += __shfl_xor(rs2, 16, 32);
    l1 = l1 * c1 + rs1;
    l2 = l2 * c2 + rs2;

    // broadcast per-row rescale factors (row q lives in lane q of the stats)
    float cr1[8], cr2[8];
#pragma unroll
    for (int j = 0; j < 8; ++j) {
      cr1[j] = __shfl(c1, j + 8 * g, 32);
      cr2[j] = __shfl(c2, j + 8 * g, 32);
    }
#pragma unroll
    for (int nt = 0; nt < 4; ++nt)
#pragma unroll
      for (int j = 0; j < 8; ++j) { o1[nt][j] *= cr1[j]; o2[nt][j] *= cr2[j]; }

    // O += P * V   (V transposed in memory: contiguous keys per channel)
#pragma unroll
    for (int nt = 0; nt < 4; ++nt) {
      const bf16* vp = Vt + (size_t)(h * 64 + nt * 16 + l15) * VT_ROW +
                       b * KEYPAD + mc * 32 + g * 8;
      v16bf bv = ld16(vp, vp + 16);
      o1[nt] = wmma_bf16(ap1, bv, o1[nt]);
      o2[nt] = wmma_bf16(ap2, bv, o2[nt]);
    }
  }

  // epilogue: normalize both softmaxes, differential combine, RMSNorm * 0.8
  float il1[8], il2[8];
#pragma unroll
  for (int j = 0; j < 8; ++j) {
    il1[j] = 1.f / __shfl(l1, j + 8 * g, 32);
    il2[j] = 1.f / __shfl(l2, j + 8 * g, 32);
  }
  float att[4][8], ssq[8];
#pragma unroll
  for (int j = 0; j < 8; ++j) ssq[j] = 0.f;
#pragma unroll
  for (int nt = 0; nt < 4; ++nt)
#pragma unroll
    for (int j = 0; j < 8; ++j) {
      float v = o1[nt][j] * il1[j] - lam * (o2[nt][j] * il2[j]);
      att[nt][j] = v;
      ssq[j] += v * v;
    }
#pragma unroll
  for (int j = 0; j < 8; ++j) {
#pragma unroll
    for (int msk = 8; msk >= 1; msk >>= 1) ssq[j] += __shfl_xor(ssq[j], msk, 32);
    ssq[j] = rsqrtf(ssq[j] * (1.f / 64.f) + 1e-5f) * 0.8f;   // rms^-1 * (1-LAMBDA_INIT)
  }
#pragma unroll
  for (int nt = 0; nt < 4; ++nt) {
    const float gch = subg[nt * 16 + l15];
#pragma unroll
    for (int j = 0; j < 8; ++j) {
      const int qr = q0 + j + 8 * g;
      if (qr < NTOK)
        Ao[(size_t)(b * NTOK + qr) * DMODEL + h * 64 + nt * 16 + l15] =
            (bf16)(att[nt][j] * ssq[j] * gch);
    }
  }
}

// ---------------- host-side orchestration ----------------
extern "C" void kernel_launch(void* const* d_in, const int* in_sizes, int n_in,
                              void* d_out, int out_size, void* d_ws, size_t ws_size,
                              hipStream_t stream) {
  (void)in_sizes; (void)n_in; (void)out_size; (void)ws_size;
  const float* x      = (const float*)d_in[0];
  const float* ln1_g  = (const float*)d_in[1];
  const float* ln1_b  = (const float*)d_in[2];
  const float* wq     = (const float*)d_in[3];
  const float* wk     = (const float*)d_in[4];
  const float* wv     = (const float*)d_in[5];
  const float* wo     = (const float*)d_in[6];
  const float* lq1    = (const float*)d_in[7];
  const float* lk1    = (const float*)d_in[8];
  const float* lq2    = (const float*)d_in[9];
  const float* lk2    = (const float*)d_in[10];
  const float* subg   = (const float*)d_in[11];
  const float* ln2_g  = (const float*)d_in[12];
  const float* ln2_b  = (const float*)d_in[13];
  const float* w1     = (const float*)d_in[14];
  const float* b1     = (const float*)d_in[15];
  const float* w2     = (const float*)d_in[16];
  const float* b2     = (const float*)d_in[17];
  float* out          = (float*)d_out;
  char*  ws           = (char*)d_ws;

  // workspace layout (bytes)
  const size_t SZ_DD = (size_t)DMODEL * DMODEL * 2;          // 1,179,648
  const size_t SZ_MD = (size_t)MLPD * DMODEL * 2;            // 4,718,592
  const size_t SZ_TD = (size_t)TOKENS * DMODEL * 2;          // 28,360,704
  const size_t SZ_VT = (size_t)DMODEL * VT_ROW * 2;          // 29,884,416
  const size_t O_WQ = 0, O_WK = O_WQ + SZ_DD, O_WV = O_WK + SZ_DD, O_WO = O_WV + SZ_DD;
  const size_t O_W1 = O_WO + SZ_DD, O_W2 = O_W1 + SZ_MD;
  const size_t O_HN = O_W2 + SZ_MD;
  const size_t O_Q  = O_HN + SZ_TD;
  const size_t O_K  = O_Q + SZ_TD;
  const size_t O_VT = O_K + SZ_TD;
  const size_t O_AO = O_VT + SZ_VT;
  const size_t O_X1 = O_AO + SZ_TD;                           // f32 residual stream

  bf16* Wqb = (bf16*)(ws + O_WQ);
  bf16* Wkb = (bf16*)(ws + O_WK);
  bf16* Wvb = (bf16*)(ws + O_WV);
  bf16* Wob = (bf16*)(ws + O_WO);
  bf16* W1b = (bf16*)(ws + O_W1);
  bf16* W2b = (bf16*)(ws + O_W2);
  bf16* Hn  = (bf16*)(ws + O_HN);
  bf16* Qb  = (bf16*)(ws + O_Q);
  bf16* Kb  = (bf16*)(ws + O_K);
  bf16* Vt  = (bf16*)(ws + O_VT);
  bf16* Ao  = (bf16*)(ws + O_AO);
  float* X1 = (float*)(ws + O_X1);
  bf16* H1  = (bf16*)(ws + O_Q);   // overlay: Q/K/Vt/Ao retired before MLP up

  const int NDD = DMODEL * DMODEL, NMD = MLPD * DMODEL;
  cvt_bf16_kernel<<<(NDD + 255) / 256, 256, 0, stream>>>(wq, Wqb, NDD);
  cvt_bf16_kernel<<<(NDD + 255) / 256, 256, 0, stream>>>(wk, Wkb, NDD);
  cvt_bf16_kernel<<<(NDD + 255) / 256, 256, 0, stream>>>(wv, Wvb, NDD);
  cvt_bf16_kernel<<<(NDD + 255) / 256, 256, 0, stream>>>(wo, Wob, NDD);
  cvt_bf16_kernel<<<(NMD + 255) / 256, 256, 0, stream>>>(w1, W1b, NMD);
  cvt_bf16_kernel<<<(NMD + 255) / 256, 256, 0, stream>>>(w2, W2b, NMD);

  const dim3 blk(256);
  const dim3 lnGrid((TOKENS + 7) / 8);
  const int  mBlk = (TOKENS + 255) / 256;   // 73

  // LN1
  layernorm_kernel<<<lnGrid, blk, 0, stream>>>(x, ln1_g, ln1_b, Hn);
  // Q (scale folded), K, V(transposed)
  wmma_gemm_kernel<<<dim3(DMODEL / 64, mBlk), blk, 0, stream>>>(
      Hn, Wqb, DMODEL, DMODEL, 0, QSCALE, nullptr, nullptr, nullptr, Qb);
  wmma_gemm_kernel<<<dim3(DMODEL / 64, mBlk), blk, 0, stream>>>(
      Hn, Wkb, DMODEL, DMODEL, 0, 1.f, nullptr, nullptr, nullptr, Kb);
  wmma_gemm_kernel<<<dim3(DMODEL / 64, mBlk), blk, 0, stream>>>(
      Hn, Wvb, DMODEL, DMODEL, 1, 1.f, nullptr, nullptr, nullptr, Vt);
  // differential attention (+ RMSNorm epilogue)
  diff_attn_kernel<<<(BATCH * NHEAD * QTILES) / 8, blk, 0, stream>>>(
      Qb, Kb, Vt, lq1, lk1, lq2, lk2, subg, Ao);
  // output projection + residual -> X1 (f32)
  wmma_gemm_kernel<<<dim3(DMODEL / 64, mBlk), blk, 0, stream>>>(
      Ao, Wob, DMODEL, DMODEL, 2, 1.f, x, nullptr, X1, nullptr);
  // LN2 (reuse Hn)
  layernorm_kernel<<<lnGrid, blk, 0, stream>>>(X1, ln2_g, ln2_b, Hn);
  // MLP up + exact GELU
  wmma_gemm_kernel<<<dim3(MLPD / 64, mBlk), blk, 0, stream>>>(
      Hn, W1b, DMODEL, MLPD, 3, 1.f, nullptr, b1, nullptr, H1);
  // MLP down + bias + residual -> out
  wmma_gemm_kernel<<<dim3(DMODEL / 64, mBlk), blk, 0, stream>>>(
      H1, W2b, MLPD, DMODEL, 4, 1.f, X1, b2, out, nullptr);
}
